// PointNetPPClassification_83296595739054
// MI455X (gfx1250) — matrix operand order repaired and checked by reading, not compiled, not run on hardware
//
#include <hip/hip_runtime.h>

// ---------------------------------------------------------------------------
// Types / helpers for CDNA5 WMMA (gfx1250, wave32)
// ---------------------------------------------------------------------------
typedef __attribute__((ext_vector_type(16))) __bf16 bf16x16;
typedef __attribute__((ext_vector_type(8)))  float  floatx8;

#define BN_SCALE 0.9999950000375f   /* 1/sqrt(1+1e-5) */

__device__ __forceinline__ unsigned short f2bf(float f) {
    union { __bf16 h; unsigned short s; } u;
    u.h = (__bf16)f;                     // native RNE convert
    return u.s;
}

union Frag {
    bf16x16 v;
    uint4   q[2];
};

// A-matrix (16x32 bf16, MxK): lane m = lane&15, half = lane>>4.
// element e -> K = e + 16*half  => one contiguous 32B run.
__device__ __forceinline__ void load_fragA(Frag& f, const unsigned short* rowBase, int half) {
    f.q[0] = *reinterpret_cast<const uint4*>(rowBase + (half << 4));
    f.q[1] = *reinterpret_cast<const uint4*>(rowBase + 8 + (half << 4));
}

// B-matrix (32x16 bf16, KxN): lane n = lane&15, half = lane>>4.
// element e -> K = e + 16*half  => one contiguous 32B run.
__device__ __forceinline__ void load_fragB(Frag& f, const unsigned short* rowBase, int half) {
    const uint4* p = reinterpret_cast<const uint4*>(rowBase + (half << 4));
    f.q[0] = p[0];
    f.q[1] = p[1];
}

__device__ __forceinline__ floatx8 wmma_bf16(const Frag& a, const Frag& b, floatx8 c) {
    return __builtin_amdgcn_wmma_f32_16x16x32_bf16(false, a.v, false, b.v,
                                                   (short)0, c, false, false);
}

// ---------------------------------------------------------------------------
// Weight pad+convert: f32 [cout][cin] -> bf16 [cout][cinp] (zero padded).
// F>0: permute columns to [features(F) | xyz(3)] layout (cin == F+3).
// ---------------------------------------------------------------------------
__global__ void padw_kernel(const float* __restrict__ src, unsigned short* __restrict__ dst,
                            int cout, int cin, int cinp, int F) {
    int i = blockIdx.x * blockDim.x + threadIdx.x;
    int tot = cout * cinp;
    if (i < tot) {
        int r = i / cinp, c = i % cinp;
        int sc = -1;
        if (F > 0) {
            if (c < F)          sc = 3 + c;      // feature columns (src cols 3..F+2)
            else if (c < F + 3) sc = c - F;      // xyz columns (src cols 0..2)
        } else if (c < cin) {
            sc = c;
        }
        dst[i] = (sc >= 0) ? f2bf(src[r * cin + sc]) : (unsigned short)0;
    }
}

// ---------------------------------------------------------------------------
// Farthest point sampling: one block per batch, coords+dists in registers,
// block argmax with jnp.argmax tie-break (smallest index wins).
// ---------------------------------------------------------------------------
template <int N, int PTS>
__global__ __launch_bounds__(256)
void fps_kernel(const float* __restrict__ xyz, int S, int* __restrict__ outIdx) {
    __shared__ float sval[256];
    __shared__ int   sidx[256];
    __shared__ int   slast;
    int b = blockIdx.x, t = threadIdx.x;
    const float* base = xyz + (size_t)b * N * 3;

    float px[PTS], py[PTS], pz[PTS], dist[PTS];
#pragma unroll
    for (int j = 0; j < PTS; ++j) {
        int i = t * PTS + j;
        px[j] = base[i * 3 + 0];
        py[j] = base[i * 3 + 1];
        pz[j] = base[i * 3 + 2];
        dist[j] = 1e10f;
    }
    if (t == 0) { slast = 0; outIdx[b * S] = 0; }
    __syncthreads();

    for (int it = 1; it < S; ++it) {
        int last = slast;
        float qx = base[last * 3 + 0], qy = base[last * 3 + 1], qz = base[last * 3 + 2];
        float best = -1.0f; int bidx = t * PTS;
#pragma unroll
        for (int j = 0; j < PTS; ++j) {
            float dx = px[j] - qx, dy = py[j] - qy, dz = pz[j] - qz;
            float d = dx * dx + dy * dy + dz * dz;
            dist[j] = fminf(dist[j], d);
            if (dist[j] > best) { best = dist[j]; bidx = t * PTS + j; }
        }
        sval[t] = best; sidx[t] = bidx;
        __syncthreads();
        for (int off = 128; off > 0; off >>= 1) {
            if (t < off) {
                float v2 = sval[t + off]; int i2 = sidx[t + off];
                if (v2 > sval[t] || (v2 == sval[t] && i2 < sidx[t])) { sval[t] = v2; sidx[t] = i2; }
            }
            __syncthreads();
        }
        if (t == 0) { slast = sidx[0]; outIdx[b * S + it] = sidx[0]; }
        __syncthreads();
    }
}

// Gather new_xyz[b][s] = xyz[b][fps_idx[b][s]]
__global__ void gather_nx_kernel(const float* __restrict__ xyz, const int* __restrict__ idx,
                                 float* __restrict__ nxyz, int S, int N, int total) {
    int i = blockIdx.x * blockDim.x + threadIdx.x;
    if (i < total) {
        int b = i / S;
        const float* src = xyz + ((size_t)b * N + idx[i]) * 3;
        nxyz[i * 3 + 0] = src[0];
        nxyz[i * 3 + 1] = src[1];
        nxyz[i * 3 + 2] = src[2];
    }
}

// ---------------------------------------------------------------------------
// Ball query: one wave per center. First 64 in-radius indices in ascending
// order; pad with the first hit (or 0). Matches reference top-k(-key) trick.
// ---------------------------------------------------------------------------
__global__ __launch_bounds__(32)
void ballq_kernel(const float* __restrict__ xyz, const float* __restrict__ nxyz,
                  int* __restrict__ nbr, int S, int N, float r2) {
    int bs = blockIdx.x;
    int b = bs / S;
    int lane = threadIdx.x;
    float cx = nxyz[bs * 3 + 0], cy = nxyz[bs * 3 + 1], cz = nxyz[bs * 3 + 2];
    const float* base = xyz + (size_t)b * N * 3;
    int* dst = nbr + (size_t)bs * 64;

    int have = 0, first = -1;
    for (int baseI = 0; baseI < N && have < 64; baseI += 32) {
        int i = baseI + lane;
        float dx = base[i * 3 + 0] - cx;
        float dy = base[i * 3 + 1] - cy;
        float dz = base[i * 3 + 2] - cz;
        float d2 = dx * dx + dy * dy + dz * dz;
        bool valid = d2 < r2;
        unsigned mask = (unsigned)__ballot(valid);
        if (first < 0 && mask != 0u) first = baseI + (__ffs(mask) - 1);
        int rank = __popc(mask & ((1u << lane) - 1u));
        if (valid && (have + rank) < 64) dst[have + rank] = i;
        have += __popc(mask);
    }
    if (first < 0) first = 0;
    if (have > 64) have = 64;
    for (int k = have + lane; k < 64; k += 32) dst[k] = first;
}

// ---------------------------------------------------------------------------
// Fused SA MLP layers (per-center, one wave, 64 samples = 4 N-tiles).
// Non-pool layer: hoist all K-chunks of B in registers per token tile.
// ---------------------------------------------------------------------------
template <int KCH>
__device__ __forceinline__ void layer_gemm(const unsigned short* __restrict__ W,
                                           const float* __restrict__ g,
                                           const float* __restrict__ bia,
                                           const unsigned short* src,   // LDS [64][KCH*32]
                                           unsigned short* dst,         // LDS [64][Cout]
                                           int Cout, int lane) {
    int half = lane >> 4, l15 = lane & 15;
    for (int nt = 0; nt < 4; ++nt) {
        int n = nt * 16 + l15;
        const unsigned short* srow = src + n * (KCH * 32);
        Frag barr[KCH];
#pragma unroll
        for (int kc = 0; kc < KCH; ++kc) load_fragB(barr[kc], srow + kc * 32, half);
        for (int mt = 0; mt < Cout / 16; ++mt) {
            const unsigned short* wrow = W + (size_t)(mt * 16 + l15) * (KCH * 32);
            __builtin_prefetch(wrow, 0, 1);
            floatx8 acc = {0.f, 0.f, 0.f, 0.f, 0.f, 0.f, 0.f, 0.f};
#pragma unroll
            for (int kc = 0; kc < KCH; ++kc) {
                Frag a;
                load_fragA(a, wrow + kc * 32, half);
                acc = wmma_bf16(a, barr[kc], acc);
            }
            union { __bf16 h[8]; uint4 q; } o;
#pragma unroll
            for (int r = 0; r < 8; ++r) {
                int c = mt * 16 + (half << 3) + r;
                float x = acc[r] * (g[c] * BN_SCALE) + bia[c];
                o.h[r] = (__bf16)(x > 0.f ? x : 0.f);
            }
            *reinterpret_cast<uint4*>(dst + n * Cout + mt * 16 + (half << 3)) = o.q;
        }
    }
}

// Pool layer: hoist A (weights) per channel tile; max over 64 samples fused.
template <int KCH>
__device__ __forceinline__ void layer_gemm_pool(const unsigned short* __restrict__ W,
                                                const float* __restrict__ g,
                                                const float* __restrict__ bia,
                                                const unsigned short* src,
                                                unsigned short* __restrict__ fout, // global bf16 [Cout]
                                                int Cout, int lane) {
    int half = lane >> 4, l15 = lane & 15;
    for (int mt = 0; mt < Cout / 16; ++mt) {
        const unsigned short* wrow = W + (size_t)(mt * 16 + l15) * (KCH * 32);
        Frag aarr[KCH];
#pragma unroll
        for (int kc = 0; kc < KCH; ++kc) load_fragA(aarr[kc], wrow + kc * 32, half);
        float sc[8], bi[8];
#pragma unroll
        for (int r = 0; r < 8; ++r) {
            int c = mt * 16 + (half << 3) + r;
            sc[r] = g[c] * BN_SCALE; bi[r] = bia[c];
        }
        float m8[8] = {0.f, 0.f, 0.f, 0.f, 0.f, 0.f, 0.f, 0.f};
        for (int nt = 0; nt < 4; ++nt) {
            floatx8 acc = {0.f, 0.f, 0.f, 0.f, 0.f, 0.f, 0.f, 0.f};
            const unsigned short* srow = src + (nt * 16 + l15) * (KCH * 32);
#pragma unroll
            for (int kc = 0; kc < KCH; ++kc) {
                Frag bf;
                load_fragB(bf, srow + kc * 32, half);
                acc = wmma_bf16(aarr[kc], bf, acc);
            }
#pragma unroll
            for (int r = 0; r < 8; ++r) {
                float x = acc[r] * sc[r] + bi[r];
                m8[r] = fmaxf(m8[r], x > 0.f ? x : 0.f);   // relu >= 0
            }
        }
#pragma unroll
        for (int r = 0; r < 8; ++r) {
            float v = m8[r];
            for (int o = 1; o < 16; o <<= 1) v = fmaxf(v, __shfl_xor(v, o, 32));
            if (l15 == 0) fout[mt * 16 + (half << 3) + r] = f2bf(v);
        }
    }
}

// Input layout (after zero fill): F==0 -> [xyz(3)|0], F>0 -> [featsBf16(F)|xyz(3)|0]
template <int CINP, int C1, int C2, int C3>
__global__ __launch_bounds__(32)
void sa_kernel(const float* __restrict__ coords, const unsigned short* __restrict__ featsBf,
               const int* __restrict__ nbr, const float* __restrict__ nxyz,
               const unsigned short* __restrict__ w1, const unsigned short* __restrict__ w2,
               const unsigned short* __restrict__ w3,
               const float* __restrict__ g1, const float* __restrict__ b1,
               const float* __restrict__ g2, const float* __restrict__ b2,
               const float* __restrict__ g3, const float* __restrict__ b3,
               unsigned short* __restrict__ fout, int S, int N, int F) {
    __shared__ alignas(16) unsigned short bufIn[64 * CINP];
    __shared__ alignas(16) unsigned short h1[64 * C1];
    __shared__ alignas(16) unsigned short h2[64 * C2];

    int bs = blockIdx.x;
    int b = bs / S;
    int lane = threadIdx.x;
    const int* nb = nbr + (size_t)bs * 64;
    float cx = nxyz[bs * 3 + 0], cy = nxyz[bs * 3 + 1], cz = nxyz[bs * 3 + 2];

    {   // zero the input staging buffer (covers channel padding)
        const uint4 z = {0u, 0u, 0u, 0u};
        for (int i = lane; i < 64 * CINP / 8; i += 32)
            reinterpret_cast<uint4*>(bufIn)[i] = z;
    }
    __syncthreads();

    // relative xyz -> channels [F, F+3)
    for (int k = lane; k < 64; k += 32) {
        int idx = nb[k];
        const float* p = coords + ((size_t)b * N + idx) * 3;
        unsigned short* row = bufIn + k * CINP + F;
        row[0] = f2bf(p[0] - cx);
        row[1] = f2bf(p[1] - cy);
        row[2] = f2bf(p[2] - cz);
    }
    // gathered bf16 features -> channels [0, F) via 16B chunks, lanes cooperate
    if (F > 0) {
        const int chunks = F >> 3;                       // 16B chunks per row
        for (int i = lane; i < 64 * chunks; i += 32) {
            int k = i / chunks, ch = i - k * chunks;
            int idx = nb[k];
            const uint4* sq = reinterpret_cast<const uint4*>(featsBf + ((size_t)b * N + idx) * F) + ch;
            *(reinterpret_cast<uint4*>(bufIn + k * CINP) + ch) = *sq;
        }
    }
    __syncthreads();
    layer_gemm<CINP / 32>(w1, g1, b1, bufIn, h1, C1, lane);
    __syncthreads();
    layer_gemm<C1 / 32>(w2, g2, b2, h1, h2, C2, lane);
    __syncthreads();
    layer_gemm_pool<C2 / 32>(w3, g3, b3, h2, fout + (size_t)bs * C3, C3, lane);
}

// ---------------------------------------------------------------------------
// SA3 prep: h0[b][t][288] = [f2(256, bf16 copy) | xyz2(3) | 0]
// ---------------------------------------------------------------------------
__global__ void sa3prep_kernel(const float* __restrict__ nx2, const unsigned short* __restrict__ f2,
                               unsigned short* __restrict__ h0, int total) {
    int i = blockIdx.x * blockDim.x + threadIdx.x;
    if (i < total) {
        int c = i % 288, bt = i / 288;
        unsigned short v = 0;
        if (c < 256)      v = f2[bt * 256 + c];
        else if (c < 259) v = f2bf(nx2[bt * 3 + (c - 256)]);
        h0[i] = v;
    }
}

// ---------------------------------------------------------------------------
// Token-major dense layer for SA3: one wave per (batch, 16-channel tile).
// in: bf16 [B][T][Kp], W: bf16 [Cout][Kp]. POOL=true fuses max over tokens.
// ---------------------------------------------------------------------------
template <bool POOL>
__global__ __launch_bounds__(32)
void dense_kernel(const unsigned short* __restrict__ in, const unsigned short* __restrict__ W,
                  const float* __restrict__ g, const float* __restrict__ bia,
                  unsigned short* __restrict__ outBf, float* __restrict__ outPool,
                  int T, int Kp, int Cout) {
    int mtiles = Cout / 16;
    int mt = blockIdx.x % mtiles;
    int b  = blockIdx.x / mtiles;
    int lane = threadIdx.x, half = lane >> 4, l15 = lane & 15;

    const unsigned short* wrow = W + (size_t)(mt * 16 + l15) * Kp;
    for (int pb = 0; pb < Kp * 2; pb += 128)
        __builtin_prefetch(reinterpret_cast<const char*>(wrow) + pb, 0, 1);

    float sc[8], bi[8];
#pragma unroll
    for (int r = 0; r < 8; ++r) {
        int c = mt * 16 + (half << 3) + r;
        sc[r] = g[c] * BN_SCALE; bi[r] = bia[c];
    }
    float m8[8] = {0.f, 0.f, 0.f, 0.f, 0.f, 0.f, 0.f, 0.f};

    for (int nt = 0; nt < T / 16; ++nt) {
        floatx8 acc = {0.f, 0.f, 0.f, 0.f, 0.f, 0.f, 0.f, 0.f};
        int t = nt * 16 + l15;
        const unsigned short* srow = in + ((size_t)b * T + t) * Kp;
        for (int kc = 0; kc < Kp / 32; ++kc) {
            Frag a, bf;
            load_fragA(a, wrow + kc * 32, half);
            load_fragB(bf, srow + kc * 32, half);
            acc = wmma_bf16(a, bf, acc);
        }
        if (POOL) {
#pragma unroll
            for (int r = 0; r < 8; ++r) {
                float x = acc[r] * sc[r] + bi[r];
                m8[r] = fmaxf(m8[r], x > 0.f ? x : 0.f);
            }
        } else {
            union { __bf16 h[8]; uint4 q; } o;
#pragma unroll
            for (int r = 0; r < 8; ++r) {
                float x = acc[r] * sc[r] + bi[r];
                o.h[r] = (__bf16)(x > 0.f ? x : 0.f);
            }
            *reinterpret_cast<uint4*>(outBf + ((size_t)b * T + t) * Cout + mt * 16 + (half << 3)) = o.q;
        }
    }
    if (POOL) {
#pragma unroll
        for (int r = 0; r < 8; ++r) {
            float v = m8[r];
            for (int o = 1; o < 16; o <<= 1) v = fmaxf(v, __shfl_xor(v, o, 32));
            if (l15 == 0) outPool[(size_t)b * Cout + mt * 16 + (half << 3) + r] = v;
        }
    }
}

// ---------------------------------------------------------------------------
// FC head: 1024 -> 512 -> 256 -> 40 (scalar f32, tiny)
// ---------------------------------------------------------------------------
__global__ __launch_bounds__(256)
void fc_kernel(const float* __restrict__ x3,
               const float* __restrict__ w0, const float* __restrict__ g0, const float* __restrict__ c0,
               const float* __restrict__ w1, const float* __restrict__ g1, const float* __restrict__ c1,
               const float* __restrict__ w2, const float* __restrict__ fcbias,
               float* __restrict__ out) {
    __shared__ float xs[1024];
    __shared__ float y1[512];
    __shared__ float y2[256];
    int b = blockIdx.x, t = threadIdx.x;
    for (int i = t; i < 1024; i += 256) xs[i] = x3[(size_t)b * 1024 + i];
    __syncthreads();
    for (int o = t; o < 512; o += 256) {
        float acc = 0.f;
        const float* wr = w0 + (size_t)o * 1024;
        for (int i = 0; i < 1024; ++i) acc += wr[i] * xs[i];
        acc = acc * (g0[o] * BN_SCALE) + c0[o];
        y1[o] = fmaxf(acc, 0.f);
    }
    __syncthreads();
    for (int o = t; o < 256; o += 256) {
        float acc = 0.f;
        const float* wr = w1 + (size_t)o * 512;
        for (int i = 0; i < 512; ++i) acc += wr[i] * y1[i];
        acc = acc * (g1[o] * BN_SCALE) + c1[o];
        y2[o] = fmaxf(acc, 0.f);
    }
    __syncthreads();
    if (t < 40) {
        float acc = 0.f;
        const float* wr = w2 + (size_t)t * 256;
        for (int i = 0; i < 256; ++i) acc += wr[i] * y2[i];
        out[b * 40 + t] = acc + fcbias[t];
    }
}

// ---------------------------------------------------------------------------
// Host orchestration
// ---------------------------------------------------------------------------
extern "C" void kernel_launch(void* const* d_in, const int* in_sizes, int n_in,
                              void* d_out, int out_size, void* d_ws, size_t ws_size,
                              hipStream_t stream) {
    const int B = 32, N1 = 4096, S1 = 512, S2 = 128;

    const float* pc    = (const float*)d_in[0];
    const float* s1w0  = (const float*)d_in[1];
    const float* s1w1  = (const float*)d_in[2];
    const float* s1w2  = (const float*)d_in[3];
    const float* s1g0  = (const float*)d_in[4];
    const float* s1g1  = (const float*)d_in[5];
    const float* s1g2  = (const float*)d_in[6];
    const float* s1b0  = (const float*)d_in[7];
    const float* s1b1  = (const float*)d_in[8];
    const float* s1b2  = (const float*)d_in[9];
    const float* s2w0  = (const float*)d_in[10];
    const float* s2w1  = (const float*)d_in[11];
    const float* s2w2  = (const float*)d_in[12];
    const float* s2g0  = (const float*)d_in[13];
    const float* s2g1  = (const float*)d_in[14];
    const float* s2g2  = (const float*)d_in[15];
    const float* s2b0  = (const float*)d_in[16];
    const float* s2b1  = (const float*)d_in[17];
    const float* s2b2  = (const float*)d_in[18];
    const float* s3w0  = (const float*)d_in[19];
    const float* s3w1  = (const float*)d_in[20];
    const float* s3w2  = (const float*)d_in[21];
    const float* s3g0  = (const float*)d_in[22];
    const float* s3g1  = (const float*)d_in[23];
    const float* s3g2  = (const float*)d_in[24];
    const float* s3b0  = (const float*)d_in[25];
    const float* s3b1  = (const float*)d_in[26];
    const float* s3b2  = (const float*)d_in[27];
    const float* fcw0  = (const float*)d_in[28];
    const float* fcw1  = (const float*)d_in[29];
    const float* fcw2  = (const float*)d_in[30];
    const float* fcg0  = (const float*)d_in[31];
    const float* fcg1  = (const float*)d_in[32];
    const float* fcb0  = (const float*)d_in[33];
    const float* fcb1  = (const float*)d_in[34];
    const float* fcbias= (const float*)d_in[35];

    char* ws = (char*)d_ws;
    size_t off = 0;
    auto alloc = [&](size_t bytes) -> void* {
        void* p = ws + off;
        off = (off + bytes + 255) & ~(size_t)255;
        return p;
    };
    int*            fps1 = (int*)           alloc((size_t)B * S1 * 4);
    float*          nx1  = (float*)         alloc((size_t)B * S1 * 3 * 4);
    int*            nbr1 = (int*)           alloc((size_t)B * S1 * 64 * 4);
    unsigned short* f1   = (unsigned short*)alloc((size_t)B * S1 * 128 * 2);   // bf16
    int*            fps2 = (int*)           alloc((size_t)B * S2 * 4);
    float*          nx2  = (float*)         alloc((size_t)B * S2 * 3 * 4);
    int*            nbr2 = (int*)           alloc((size_t)B * S2 * 64 * 4);
    unsigned short* f2   = (unsigned short*)alloc((size_t)B * S2 * 256 * 2);   // bf16
    unsigned short* h0   = (unsigned short*)alloc((size_t)B * 128 * 288 * 2);
    unsigned short* h1   = (unsigned short*)alloc((size_t)B * 128 * 256 * 2);
    unsigned short* h2   = (unsigned short*)alloc((size_t)B * 128 * 512 * 2);
    float*          x3   = (float*)         alloc((size_t)B * 1024 * 4);
    unsigned short* w1a  = (unsigned short*)alloc((size_t)64 * 32 * 2);
    unsigned short* w1b  = (unsigned short*)alloc((size_t)64 * 64 * 2);
    unsigned short* w1c  = (unsigned short*)alloc((size_t)128 * 64 * 2);
    unsigned short* w2a  = (unsigned short*)alloc((size_t)128 * 160 * 2);
    unsigned short* w2b  = (unsigned short*)alloc((size_t)128 * 128 * 2);
    unsigned short* w2c  = (unsigned short*)alloc((size_t)256 * 128 * 2);
    unsigned short* w3a  = (unsigned short*)alloc((size_t)256 * 288 * 2);
    unsigned short* w3b  = (unsigned short*)alloc((size_t)512 * 256 * 2);
    unsigned short* w3c  = (unsigned short*)alloc((size_t)1024 * 512 * 2);

    auto padw = [&](const float* src, unsigned short* dst, int cout, int cin, int cinp, int F) {
        int tot = cout * cinp;
        padw_kernel<<<(tot + 255) / 256, 256, 0, stream>>>(src, dst, cout, cin, cinp, F);
    };
    padw(s1w0, w1a,   64,   3,  32, 0);
    padw(s1w1, w1b,   64,  64,  64, 0);
    padw(s1w2, w1c,  128,  64,  64, 0);
    padw(s2w0, w2a,  128, 131, 160, 128);  // [f1(128)|xyz(3)] column order
    padw(s2w1, w2b,  128, 128, 128, 0);
    padw(s2w2, w2c,  256, 128, 128, 0);
    padw(s3w0, w3a,  256, 259, 288, 256);  // [f2(256)|xyz(3)] column order
    padw(s3w1, w3b,  512, 256, 256, 0);
    padw(s3w2, w3c, 1024, 512, 512, 0);

    // ---- SA1 ----
    fps_kernel<4096, 16><<<B, 256, 0, stream>>>(pc, S1, fps1);
    gather_nx_kernel<<<(B * S1 + 255) / 256, 256, 0, stream>>>(pc, fps1, nx1, S1, N1, B * S1);
    ballq_kernel<<<B * S1, 32, 0, stream>>>(pc, nx1, nbr1, S1, N1, 0.2f * 0.2f);
    sa_kernel<32, 64, 64, 128><<<B * S1, 32, 0, stream>>>(
        pc, nullptr, nbr1, nx1, w1a, w1b, w1c,
        s1g0, s1b0, s1g1, s1b1, s1g2, s1b2, f1, S1, N1, 0);

    // ---- SA2 ----
    fps_kernel<512, 2><<<B, 256, 0, stream>>>(nx1, S2, fps2);
    gather_nx_kernel<<<(B * S2 + 255) / 256, 256, 0, stream>>>(nx1, fps2, nx2, S2, S1, B * S2);
    ballq_kernel<<<B * S2, 32, 0, stream>>>(nx1, nx2, nbr2, S2, S1, 0.4f * 0.4f);
    sa_kernel<160, 128, 128, 256><<<B * S2, 32, 0, stream>>>(
        nx1, f1, nbr2, nx2, w2a, w2b, w2c,
        s2g0, s2b0, s2g1, s2b1, s2g2, s2b2, f2, S2, S1, 128);

    // ---- SA3 (global MLP over 128 tokens) ----
    {
        int tot = B * 128 * 288;
        sa3prep_kernel<<<(tot + 255) / 256, 256, 0, stream>>>(nx2, f2, h0, tot);
    }
    dense_kernel<false><<<B * (256 / 16),  32, 0, stream>>>(h0, w3a, s3g0, s3b0, h1, nullptr, 128, 288, 256);
    dense_kernel<false><<<B * (512 / 16),  32, 0, stream>>>(h1, w3b, s3g1, s3b1, h2, nullptr, 128, 256, 512);
    dense_kernel<true> <<<B * (1024 / 16), 32, 0, stream>>>(h2, w3c, s3g2, s3b2, nullptr, x3, 128, 512, 1024);

    // ---- FC head ----
    fc_kernel<<<B, 256, 0, stream>>>(x3, fcw0, fcg0, fcb0, fcw1, fcg1, fcb1, fcw2, fcbias,
                                     (float*)d_out);

    (void)in_sizes; (void)n_in; (void)out_size; (void)ws_size;
}
